// GatedSSM_74818330296397
// MI455X (gfx1250) — matrix operand order repaired and checked by reading, not compiled, not run on hardware
//
#include <hip/hip_runtime.h>
#include <cstdint>
#include <cstddef>

// ---------------- types ----------------
typedef _Float16 h8   __attribute__((ext_vector_type(8)));
typedef _Float16 v16h __attribute__((ext_vector_type(16)));
typedef float    v8f  __attribute__((ext_vector_type(8)));

#define DIMS  1024
#define STATE 2048
#define BATCH 4
#define SEQ   2048
#define ROWS  (BATCH*SEQ)   // 8192 tokens
#define NCOMB (4*STATE)     // 8192 = STATE (K) + 3*STATE (u,g_in,g_out)

__device__ __forceinline__ float sigmoidf_(float x) {
    return 1.0f / (1.0f + __expf(-x));
}

// ---------------- 1) split RMS norm -> f16 ----------------
// one block (256 threads) per token row
__global__ __launch_bounds__(256)
void norm_kernel(const float* __restrict__ x,
                 const float* __restrict__ ls, const float* __restrict__ rs,
                 const float* __restrict__ ss, _Float16* __restrict__ xn)
{
    int row = blockIdx.x;
    int t   = threadIdx.x;                 // 0..255
    const float* xr = x + (size_t)row * DIMS;

    float l0 = xr[t],       l1 = xr[t + 256];
    float r0 = xr[512 + t], r1 = xr[768 + t];

    __shared__ float redL[256], redR[256];
    redL[t] = l0 * l0 + l1 * l1;
    redR[t] = r0 * r0 + r1 * r1;
    __syncthreads();
    for (int s = 128; s > 0; s >>= 1) {
        if (t < s) { redL[t] += redL[t + s]; redR[t] += redR[t + s]; }
        __syncthreads();
    }
    __shared__ float nl, nr;
    if (t == 0) {
        nl = 1.0f / (sqrtf(redL[0] * (1.0f / 512.0f)) + 1e-8f);
        nr = 1.0f / (sqrtf(redR[0] * (1.0f / 512.0f)) + 1e-8f);
    }
    __syncthreads();

    _Float16* o = xn + (size_t)row * DIMS;
    o[t]       = (_Float16)(l0 * ls[t]       * nl * ss[t]);
    o[t + 256] = (_Float16)(l1 * ls[t + 256] * nl * ss[t + 256]);
    o[512 + t] = (_Float16)(r0 * rs[t]       * nr * ss[512 + t]);
    o[768 + t] = (_Float16)(r1 * rs[t + 256] * nr * ss[768 + t]);
}

// ---------------- 2) weight packing fp32 -> f16 ----------------
__global__ __launch_bounds__(256)
void pack_wcomb(const float* __restrict__ Wk, const float* __restrict__ Wugg,
                _Float16* __restrict__ Wc)
{
    size_t i = (size_t)blockIdx.x * 256 + threadIdx.x;   // over 1024*8192
    int col = (int)(i & (NCOMB - 1));
    int r   = (int)(i >> 13);
    float v = (col < STATE) ? Wk[(size_t)r * STATE + col]
                            : Wugg[(size_t)r * (3 * STATE) + (col - STATE)];
    Wc[i] = (_Float16)v;
}

__global__ __launch_bounds__(256)
void pack_wout(const float* __restrict__ W, _Float16* __restrict__ Wo)
{
    size_t i = (size_t)blockIdx.x * 256 + threadIdx.x;   // over 2048*1024
    Wo[i] = (_Float16)W[i];
}

// ---------------- 3/6) WMMA GEMM: C[M,N] = A[M,K] * B[K,N] (+Res) ----------------
// Block tile 128x256, 8 waves (2 along M x 4 along N), wave tile 64x64.
// Double-buffered LDS, one barrier per K-step, 16 WMMAs per wave per K-step.
#define BM 128
#define BN 256
#define BK 32
#define LSTR 40   // padded LDS row stride in halves (80B, keeps 16B alignment)

__device__ __forceinline__ v16h mkfrag(h8 lo, h8 hi)
{
    v16h r;
#pragma unroll
    for (int i = 0; i < 8; i++) { r[i] = lo[i]; r[8 + i] = hi[i]; }
    return r;
}

template <bool HAS_RES>
__global__ __launch_bounds__(256)
void wmma_gemm(const _Float16* __restrict__ A, const _Float16* __restrict__ B,
               float* __restrict__ C, const float* __restrict__ Res,
               int M, int N, int K)
{
    __shared__ _Float16 smA[2 * BM * LSTR];   // 20 KB
    __shared__ _Float16 smB[2 * BN * LSTR];   // 40 KB

    int tid  = threadIdx.x;
    int lane = tid & 31;
    int wid  = tid >> 5;
    int wm   = wid & 1;        // 2 waves along M  (64 rows each)
    int wn   = wid >> 1;       // 4 waves along N  (64 cols each)
    int mblk = blockIdx.y * BM;
    int nblk = blockIdx.x * BN;

    int lm    = lane & 15;
    int lksel = lane >> 4;

    v8f zero = {};
    v8f acc[4][4];
#pragma unroll
    for (int f = 0; f < 4; f++)
#pragma unroll
        for (int g = 0; g < 4; g++) acc[f][g] = zero;

    h8 ra[2];   // A staging: 2 chunks of 8 halves per thread
    h8 rb[4];   // B staging: 4 chunks of 8 halves per thread

    // ---- global -> regs for tile at k0 ----
    auto load_regs = [&](int k0) {
#pragma unroll
        for (int j = 0; j < 2; j++) {
            int c    = tid * 2 + j;          // 0..511 : A tile 128x32
            int row  = c >> 2;
            int koff = (c & 3) * 8;
            ra[j] = *(const h8*)(A + (size_t)(mblk + row) * K + (k0 + koff));
        }
#pragma unroll
        for (int j = 0; j < 4; j++) {
            int c    = tid * 4 + j;          // 0..1023 : B tile 32x256
            int krow = c >> 5;
            int noff = (c & 31) * 8;
            rb[j] = *(const h8*)(B + (size_t)(k0 + krow) * N + (nblk + noff));
        }
    };

    // ---- regs -> LDS buffer `buf` (B stored transposed [n][k]) ----
    auto store_lds = [&](int buf) {
        _Float16* sA = smA + buf * (BM * LSTR);
        _Float16* sB = smB + buf * (BN * LSTR);
#pragma unroll
        for (int j = 0; j < 2; j++) {
            int c    = tid * 2 + j;
            int row  = c >> 2;
            int koff = (c & 3) * 8;
            *(h8*)(&sA[row * LSTR + koff]) = ra[j];
        }
#pragma unroll
        for (int j = 0; j < 4; j++) {
            int c    = tid * 4 + j;
            int krow = c >> 5;
            int noff = (c & 31) * 8;
#pragma unroll
            for (int e = 0; e < 8; e++)
                sB[(noff + e) * LSTR + krow] = rb[j][e];
        }
    };

    int nk = K / BK;
    load_regs(0);
    store_lds(0);
    __syncthreads();

    for (int i = 0; i < nk; i++) {
        int cur = i & 1;
        if (i + 1 < nk) load_regs((i + 1) * BK);

        const _Float16* sA = smA + cur * (BM * LSTR);
        const _Float16* sB = smB + cur * (BN * LSTR);

        // fragments per ISA layout
        v16h afrag[4];
#pragma unroll
        for (int f = 0; f < 4; f++) {
            int row = wm * 64 + f * 16 + lm;
            int kb  = lksel * 8;             // lanes0-15: K0-7/16-23; lanes16-31: K8-15/24-31
            h8 lo = *(const h8*)(&sA[row * LSTR + kb]);
            h8 hi = *(const h8*)(&sA[row * LSTR + kb + 16]);
            afrag[f] = mkfrag(lo, hi);
        }
        v16h bfrag[4];
#pragma unroll
        for (int g = 0; g < 4; g++) {
            int col = wn * 64 + g * 16 + lm;
            int kb  = lksel * 8;             // lanes0-15: K0-7/16-23; lanes16-31: K8-15/24-31
            h8 lo = *(const h8*)(&sB[col * LSTR + kb]);
            h8 hi = *(const h8*)(&sB[col * LSTR + kb + 16]);
            bfrag[g] = mkfrag(lo, hi);
        }

#pragma unroll
        for (int f = 0; f < 4; f++)
#pragma unroll
            for (int g = 0; g < 4; g++)
                acc[f][g] = __builtin_amdgcn_wmma_f32_16x16x32_f16(
                    false, afrag[f], false, bfrag[g],
                    (short)0, acc[f][g], false, false);

        if (i + 1 < nk) {
            store_lds((i + 1) & 1);
            __syncthreads();
        }
    }

    // --- epilogue: C/D layout — VGPR r -> M = r + 8*(lane/16), N = lane%16 ---
    int cn_l   = lane & 15;
    int cm_off = (lane >> 4) * 8;
#pragma unroll
    for (int f = 0; f < 4; f++) {
        int m0 = mblk + wm * 64 + f * 16 + cm_off;
#pragma unroll
        for (int g = 0; g < 4; g++) {
            int n0 = nblk + wn * 64 + g * 16 + cn_l;
#pragma unroll
            for (int r = 0; r < 8; r++) {
                size_t idx = (size_t)(m0 + r) * N + n0;
                float v = acc[f][g][r];
                if constexpr (HAS_RES) v += Res[idx];
                C[idx] = v;
            }
        }
    }
}

// ---------------- 4) activations in place on KUGG ----------------
// KUGG row layout: [K | u | g_in | g_out], row stride NCOMB
__global__ __launch_bounds__(256)
void act_kernel(float* __restrict__ KUGG)
{
    size_t i  = (size_t)blockIdx.x * 256 + threadIdx.x;  // over ROWS*STATE
    size_t rw = i >> 11;
    int    j  = (int)(i & (STATE - 1));
    float* r  = KUGG + rw * NCOMB;
    float Ks = sigmoidf_(r[j]);
    float up = r[STATE + j] * sigmoidf_(r[2 * STATE + j]) * (1.0f - Ks);
    r[j]         = Ks;
    r[STATE + j] = up;
}

// ---------------- 5) linear recurrence scan ----------------
// h_t = K_{t-1} * h_{t-1} + u_t ; out_t = h_t * sigmoid(g_out_t)  -> f16
__global__ __launch_bounds__(256)
void scan_kernel(const float* __restrict__ KUGG, _Float16* __restrict__ out16)
{
    int idx = blockIdx.x * 256 + threadIdx.x;     // 0..BATCH*STATE-1
    int b = idx >> 11;
    int n = idx & (STATE - 1);
    const float* base = KUGG + (size_t)b * SEQ * NCOMB;

    float h = 0.0f, kprev = 1.0f;
    for (int t = 0; t < SEQ; t++) {
        const float* row = base + (size_t)t * NCOMB;
        float Kt = row[n];               // sigmoided
        float ut = row[STATE + n];       // gated u'
        float go = row[3 * STATE + n];   // raw g_out
        h = kprev * h + ut;
        kprev = Kt;
        out16[((size_t)(b * SEQ + t)) * STATE + n] = (_Float16)(h * sigmoidf_(go));
    }
}

// ---------------- launch ----------------
extern "C" void kernel_launch(void* const* d_in, const int* in_sizes, int n_in,
                              void* d_out, int out_size, void* d_ws, size_t ws_size,
                              hipStream_t stream)
{
    const float* x    = (const float*)d_in[0];
    const float* ls   = (const float*)d_in[1];
    const float* rs   = (const float*)d_in[2];
    const float* ss   = (const float*)d_in[3];
    const float* Wk   = (const float*)d_in[4];
    const float* Wugg = (const float*)d_in[5];
    const float* Wout = (const float*)d_in[6];
    float* y = (float*)d_out;

    char* ws = (char*)d_ws;
    _Float16* XN16  = (_Float16*)(ws);                                // 16 MB
    _Float16* WC16  = (_Float16*)(ws + (size_t)16777216);             // 16 MB
    _Float16* WO16  = (_Float16*)(ws + (size_t)33554432);             //  4 MB
    float*    KUGG  = (float*)   (ws + (size_t)37748736);             // 256 MB
    _Float16* OUT16 = (_Float16*)(ws + (size_t)37748736 + 268435456); // 32 MB

    norm_kernel<<<ROWS, 256, 0, stream>>>(x, ls, rs, ss, XN16);
    pack_wcomb<<<(DIMS * NCOMB) / 256, 256, 0, stream>>>(Wk, Wugg, WC16);
    pack_wout<<<(STATE * DIMS) / 256, 256, 0, stream>>>(Wout, WO16);

    // GEMM1: [8192,1024] x [1024,8192] -> KUGG
    wmma_gemm<false><<<dim3(NCOMB / BN, ROWS / BM), 256, 0, stream>>>(
        XN16, WC16, KUGG, nullptr, ROWS, NCOMB, DIMS);

    act_kernel<<<(ROWS * STATE) / 256, 256, 0, stream>>>(KUGG);
    scan_kernel<<<(BATCH * STATE) / 256, 256, 0, stream>>>(KUGG, OUT16);

    // GEMM2: [8192,2048] x [2048,1024] + residual -> y
    wmma_gemm<true><<<dim3(DIMS / BN, ROWS / BM), 256, 0, stream>>>(
        OUT16, WO16, y, x, ROWS, DIMS, STATE);
}
